// CoordinateRefiner_75222057222743
// MI455X (gfx1250) — matrix operand order repaired and checked by reading, not compiled, not run on hardware
//
#include <hip/hip_runtime.h>

// ---------------- CDNA5 WMMA types & helpers ----------------
typedef __attribute__((ext_vector_type(16))) __bf16 v16bf;
typedef __attribute__((ext_vector_type(8)))  float  v8f;
typedef __attribute__((ext_vector_type(4)))  unsigned int v4u;
typedef __attribute__((ext_vector_type(8)))  int v8i;
typedef __attribute__((ext_vector_type(4)))  int v4i;
typedef unsigned short ushort_t;

#define NB    4
#define LSEQ  512
#define NNODE 2048      // B*L
#define CC    128
#define HH    4
#define DHH   32
#define SD    640
#define EMAX  131072
#define NLAY  3

__device__ __forceinline__ unsigned bf16_1(float f) {
  unsigned u = __float_as_uint(f);
  u += 0x7FFFu + ((u >> 16) & 1u);       // round-to-nearest-even
  return u >> 16;
}
__device__ __forceinline__ unsigned bf16_2(float a, float b) {
  return bf16_1(a) | (bf16_1(b) << 16);
}

__device__ __forceinline__ v8f zero8() {
  v8f z = {0.f, 0.f, 0.f, 0.f, 0.f, 0.f, 0.f, 0.f};
  return z;
}

__device__ __forceinline__ v8f wmma_bf16(v16bf a, v16bf b, v8f c) {
  // (neg_a, A, neg_b, B, c_mod, C, reuse_a, reuse_b)
  return __builtin_amdgcn_wmma_f32_16x16x32_bf16(false, a, false, b, (short)0, c,
                                                 false, false);
}

// A operand (16x32 bf16, MxK): lane m=L&15, hi=L>>4.
// v0..3 hold K = hi*8 + {0..7}; v4..7 hold K = 16 + hi*8 + {0..7}.
__device__ __forceinline__ v16bf fragA_bf16(const ushort_t* p, int hi) {
  union { int4 i[2]; v16bf v; } u;
  u.i[0] = *reinterpret_cast<const int4*>(p + hi * 8);
  u.i[1] = *reinterpret_cast<const int4*>(p + 16 + hi * 8);
  return u.v;
}
// B operand (32x16 bf16, KxN): lane n=L&15, hi=L>>4; v0..3 hold K = hi*8 + {0..7}, v4..7 hold K = 16 + hi*8 + {0..7}.
// Weights stored N-major: row n holds K consecutively.
__device__ __forceinline__ v16bf fragB_bf16(const ushort_t* p, int hi) {
  union { int4 i[2]; v16bf v; } u;
  u.i[0] = *reinterpret_cast<const int4*>(p + hi * 8);
  u.i[1] = *reinterpret_cast<const int4*>(p + 16 + hi * 8);
  return u.v;
}

// ---------------- Tensor Data Mover: 2D bf16 tile Global -> LDS ----------------
// D# per CDNA5 ISA. data_size=1 (2 bytes/elem).
// Copies tile_rows rows x tile_k elements, row stride = row_stride elements,
// packed contiguously into LDS at lds_off.
// 6-arg builtin form (clang-23 / therock headers): g0, g1, g2, g3, ext, cpol.
__device__ __forceinline__ void tdm_load_2d(unsigned lds_off, const void* gaddr,
                                            unsigned tile_k, unsigned tile_rows,
                                            unsigned row_stride) {
  unsigned long long ga = (unsigned long long)(size_t)gaddr;
  v4u g0;
  g0[0] = 1u;                                          // count=1, user descriptor
  g0[1] = lds_off;                                     // lds_addr (bytes)
  g0[2] = (unsigned)(ga & 0xFFFFFFFFu);                // global_addr[31:0]
  g0[3] = (unsigned)((ga >> 32) & 0x01FFFFFFu) | (2u << 30);  // addr[56:32], type=2
  v8i g1;
  g1[0] = (int)(1u << 16);                             // wg_mask=0, data_size=1 (2B)
  g1[1] = (int)((row_stride & 0xFFFFu) << 16);         // tensor_dim0[15:0]
  g1[2] = (int)(((row_stride >> 16) & 0xFFFFu) | ((tile_rows & 0xFFFFu) << 16));
  g1[3] = (int)(((tile_rows >> 16) & 0xFFFFu) | ((tile_k & 0xFFFFu) << 16)); // tile_dim0
  g1[4] = (int)(tile_rows & 0xFFFFu);                  // tile_dim1 (tile_dim2=0)
  g1[5] = (int)row_stride;                             // tensor_dim0_stride[31:0]
  g1[6] = 0;
  g1[7] = 0;
  v4i gz4 = {0, 0, 0, 0};
  v8i gz8 = {0, 0, 0, 0, 0, 0, 0, 0};
  __builtin_amdgcn_tensor_load_to_lds(g0, g1, gz4, gz4, gz8, 0);
}

// ---------------- one-time conversions ----------------
// f32 [K][128] weight -> bf16 [128][K] (N-major, B-operand layout)
__global__ void k_w2bf(const float* __restrict__ W, ushort_t* __restrict__ out, int K) {
  int i = blockIdx.x * 256 + threadIdx.x;              // over K*128, coalesced reads
  if (i < K * 128) {
    int k = i >> 7, n = i & 127;
    out[(size_t)n * K + k] = (ushort_t)bf16_1(W[i]);
  }
}
// plain f32 -> bf16 elementwise
__global__ void k_f2bf(const float* __restrict__ A, ushort_t* __restrict__ out, int n) {
  int i = blockIdx.x * 256 + threadIdx.x;
  if (i < n) out[i] = (ushort_t)bf16_1(A[i]);
}

// ---------------- edge feature build (once): pair_rep gather -> bf16 ----------------
__global__ void k_edgefeat(const float* __restrict__ pair, const float* __restrict__ bppm,
                           const int* __restrict__ src, const int* __restrict__ dst,
                           ushort_t* __restrict__ ebf, float* __restrict__ bpe) {
  int e = blockIdx.x * 8 + (threadIdx.x >> 5);
  int lane = threadIdx.x & 31;
  int s = src[e], d = dst[e];
  int b = s >> 9;                // /L
  int j = d - (b << 9);
  size_t po = ((size_t)s * LSEQ + (size_t)j) * CC;
  float4 v = *reinterpret_cast<const float4*>(pair + po + lane * 4);
  uint2 o;
  o.x = bf16_2(v.x, v.y);
  o.y = bf16_2(v.z, v.w);
  *reinterpret_cast<uint2*>(ebf + (size_t)e * CC + lane * 4) = o;
  if (lane == 0) bpe[e] = bppm[(size_t)s * LSEQ + j];
}

// ---------------- generic GEMM: Out[M,128] = (relu?)(A[M,K] @ W[K,128]) ----------------
// A: bf16 [M][K]; Wb: bf16 [128][K] (N-major). block = 128 threads (4 waves);
// each wave owns a 16-row tile across all 128 output cols. W chunk staged by TDM.
__global__ void k_gemm(const ushort_t* __restrict__ A, const ushort_t* __restrict__ Wb,
                       float* __restrict__ Out, int M, int K, int do_relu) {
  __shared__ ushort_t Blds[128 * 32];   // [n][k-chunk] bf16, 8 KB
  int tid = threadIdx.x;
  int wid = tid >> 5, lane = tid & 31;
  int n15 = lane & 15, hi = lane >> 4;
  int row0 = (blockIdx.x * 4 + wid) * 16;

  v8f acc[8];
#pragma unroll
  for (int t = 0; t < 8; ++t) acc[t] = zero8();

  int nchunks = K >> 5;
  for (int kc = 0; kc < nchunks; ++kc) {
    __syncthreads();
    if (tid == 0) {
      // DMA the [128 rows x 32 k] bf16 tile (row stride K) into LDS
      tdm_load_2d((unsigned)(size_t)Blds, Wb + kc * 32, 32u, 128u, (unsigned)K);
      __builtin_amdgcn_s_wait_tensorcnt(0);
    }
    __syncthreads();
    const ushort_t* arow = A + (size_t)(row0 + n15) * K + kc * 32;
    v16bf a = fragA_bf16(arow, hi);
#pragma unroll
    for (int t = 0; t < 8; ++t)
      acc[t] = wmma_bf16(a, fragB_bf16(&Blds[(t * 16 + n15) * 32], hi), acc[t]);
  }
#pragma unroll
  for (int t = 0; t < 8; ++t) {
#pragma unroll
    for (int r = 0; r < 8; ++r) {
      float v = acc[t][r];
      if (do_relu) v = fmaxf(v, 0.f);
      Out[(size_t)(row0 + r + 8 * hi) * 128 + t * 16 + n15] = v;
    }
  }
}

// ---------------- fused edge kernel: WMMA eb + logits + tanh-dot ----------------
// block = 256 threads (8 waves); each wave: 16-edge tiles; We (bf16, N-major) via TDM.
__global__ void k_edge(const ushort_t* __restrict__ ebf, const float* __restrict__ bpe,
                       const float* __restrict__ hq, const float* __restrict__ hk,
                       const float* __restrict__ x,
                       const int* __restrict__ src, const int* __restrict__ dst,
                       const float* __restrict__ emask,
                       const ushort_t* __restrict__ Web,  // bf16 [128][128] N-major
                       const float* __restrict__ We_l,    // f32 [129][128] (row 128 used)
                       const float* __restrict__ wd_l,    // [128]
                       const float* __restrict__ wx_l,    // [128]
                       float* __restrict__ logits,        // [E][4]
                       float* __restrict__ sdot) {        // [E]
  __shared__ ushort_t Welds[128 * 128];  // [n][k] bf16, 32 KB
  __shared__ float webl[128], wdl[128], wxl[128];
  int tid = threadIdx.x;
  if (tid == 0) {
    tdm_load_2d((unsigned)(size_t)Welds, Web, 128u, 128u, 128u);
    __builtin_amdgcn_s_wait_tensorcnt(0);
  }
  for (int i = tid; i < 128; i += 256) {
    webl[i] = We_l[128 * 128 + i];       // bppm weight row
    wdl[i]  = wd_l[i];
    wxl[i]  = wx_l[i];
  }
  __syncthreads();

  int wid = tid >> 5, lane = tid & 31;
  int n15 = lane & 15, hi = lane >> 4;
  const float inv = 0.17677669529663687f;  // 1/sqrt(32)

  for (int it = 0; it < 4; ++it) {
    int tile = it * (gridDim.x * 8) + blockIdx.x * 8 + wid;
    int base = tile * 16;
    // per-lane scalars for edge (base + n15); lanes 16..31 mirror 0..15
    int em = base + n15;
    int sm = src[em], dm = dst[em];
    float bpm = bpe[em];
    float rx = x[sm * 3 + 0] - x[dm * 3 + 0];
    float ry = x[sm * 3 + 1] - x[dm * 3 + 1];
    float rz = x[sm * 3 + 2] - x[dm * 3 + 2];
    float d2m = rx * rx + ry * ry + rz * rz;

    const ushort_t* arow = ebf + (size_t)em * 128;
    v16bf a0 = fragA_bf16(arow + 0,  hi);
    v16bf a1 = fragA_bf16(arow + 32, hi);
    v16bf a2 = fragA_bf16(arow + 64, hi);
    v16bf a3 = fragA_bf16(arow + 96, hi);
    if (it < 3) {  // prefetch next tile's edge-feature row for this lane
      __builtin_prefetch(ebf + (size_t)(em + 2048 * 16) * 128, 0, 1);
    }

    // broadcast per-edge scalars for this lane's 8 accumulator rows (edges hi*8+r)
    int seA[8], deA[8];
    float d2A[8], bpA[8];
#pragma unroll
    for (int r = 0; r < 8; ++r) {
      int es = hi * 8 + r;
      seA[r] = __shfl(sm, es, 32);
      deA[r] = __shfl(dm, es, 32);
      d2A[r] = __shfl(d2m, es, 32);
      bpA[r] = __shfl(bpm, es, 32);
    }

    float lgp[4][8], sdp[8];
#pragma unroll
    for (int h = 0; h < 4; ++h)
#pragma unroll
      for (int r = 0; r < 8; ++r) lgp[h][r] = 0.f;
#pragma unroll
    for (int r = 0; r < 8; ++r) sdp[r] = 0.f;

#pragma unroll
    for (int t = 0; t < 8; ++t) {
      const ushort_t* brow = &Welds[(t * 16 + n15) * 128];
      v8f acc = zero8();
      acc = wmma_bf16(a0, fragB_bf16(brow + 0,  hi), acc);
      acc = wmma_bf16(a1, fragB_bf16(brow + 32, hi), acc);
      acc = wmma_bf16(a2, fragB_bf16(brow + 64, hi), acc);
      acc = wmma_bf16(a3, fragB_bf16(brow + 96, hi), acc);
      int c = t * 16 + n15;
      float web_c = webl[c], wd_c = wdl[c], wx_c = wxl[c];
#pragma unroll
      for (int r = 0; r < 8; ++r) {
        float eb = acc[r] + bpA[r] * web_c + d2A[r] * wd_c;   // + bppm*We[128] + d2*wd
        eb = fmaxf(eb, 0.f);
        float keb = hk[(size_t)seA[r] * 128 + c] + eb;        // k + eb
        lgp[t >> 1][r] = fmaf(hq[(size_t)deA[r] * 128 + c], keb, lgp[t >> 1][r]);
        sdp[r] = fmaf(keb, wx_c, sdp[r]);
      }
    }
    // reduce partial dot-products across the 16 lanes of each half-wave
#pragma unroll
    for (int h = 0; h < 4; ++h)
#pragma unroll
      for (int r = 0; r < 8; ++r) {
#pragma unroll
        for (int m = 1; m <= 8; m <<= 1) lgp[h][r] += __shfl_xor(lgp[h][r], m, 32);
      }
#pragma unroll
    for (int r = 0; r < 8; ++r) {
#pragma unroll
      for (int m = 1; m <= 8; m <<= 1) sdp[r] += __shfl_xor(sdp[r], m, 32);
    }
#pragma unroll
    for (int r = 0; r < 8; ++r) {
      if (n15 == r) {
        int ee = base + hi * 8 + r;
        float bias = (emask[ee] - 1.f) * 1e9f;
        logits[ee * 4 + 0] = lgp[0][r] * inv + bias;
        logits[ee * 4 + 1] = lgp[1][r] * inv + bias;
        logits[ee * 4 + 2] = lgp[2][r] * inv + bias;
        logits[ee * 4 + 3] = lgp[3][r] * inv + bias;
        sdot[ee] = tanhf(sdp[r]);
      }
    }
  }
}

// ---------------- per-layer init of reduction buffers ----------------
__global__ void k_init(float* den, float* agg, float* dx, unsigned* nmax) {
  int i = blockIdx.x * 256 + threadIdx.x;
  if (i < NNODE * CC) agg[i] = 0.f;
  if (i < NNODE * HH) { den[i] = 0.f; nmax[i] = 0u; }
  if (i < NNODE * 3) dx[i] = 0.f;
}

// ---------------- segment softmax pass 1: atomic max (monotonic uint key) --------
__global__ void k_smax(const float* __restrict__ logits, const int* __restrict__ dst,
                       unsigned* __restrict__ nmax) {
  int idx = blockIdx.x * 256 + threadIdx.x;        // E*4
  int e = idx >> 2, h = idx & 3;
  int d = dst[e];
  unsigned u = __float_as_uint(logits[idx]);
  unsigned key = (u & 0x80000000u) ? ~u : (u | 0x80000000u);
  atomicMax(&nmax[d * 4 + h], key);
}

// ---------------- pass 2: exp + atomic denominator ----------------
__global__ void k_sexp(const float* __restrict__ logits, const int* __restrict__ dst,
                       const unsigned* __restrict__ nmax, float* __restrict__ aex,
                       float* __restrict__ den) {
  int idx = blockIdx.x * 256 + threadIdx.x;
  int e = idx >> 2, h = idx & 3;
  int d = dst[e];
  unsigned key = nmax[d * 4 + h];
  float m = 0.f;
  if (key != 0u) {
    unsigned u = (key & 0x80000000u) ? (key & 0x7FFFFFFFu) : ~key;
    m = __uint_as_float(u);
  }
  float ex = __expf(logits[idx] - m);
  aex[idx] = ex;
  atomicAdd(&den[d * 4 + h], ex);
}

// ---------------- pass 3: message + coordinate-force aggregation ----------------
__global__ void k_agg(const float* __restrict__ aex, const float* __restrict__ den,
                      const float* __restrict__ hv, const float* __restrict__ sdot,
                      const float* __restrict__ x, const int* __restrict__ src,
                      const int* __restrict__ dst, const float* __restrict__ emask,
                      float* __restrict__ agg, float* __restrict__ dx) {
  int e = blockIdx.x * 8 + (threadIdx.x >> 5);
  int lane = threadIdx.x & 31;
  int s = src[e], d = dst[e];
  float mk = emask[e];
  int h = lane >> 3;                               // 4 channels per lane
  float alpha = aex[e * 4 + h] / (den[d * 4 + h] + 1e-9f) * mk;
  float4 v = *reinterpret_cast<const float4*>(hv + (size_t)s * CC + lane * 4);
  float* ap = agg + (size_t)d * CC + lane * 4;
  atomicAdd(ap + 0, alpha * v.x);
  atomicAdd(ap + 1, alpha * v.y);
  atomicAdd(ap + 2, alpha * v.z);
  atomicAdd(ap + 3, alpha * v.w);
  if (lane == 0) {
    float am = 0.f;
#pragma unroll
    for (int h2 = 0; h2 < 4; ++h2) am += aex[e * 4 + h2] / (den[d * 4 + h2] + 1e-9f);
    am = am * 0.25f * mk;
    float sv = sdot[e] * am * mk;
    float rx = x[s * 3 + 0] - x[d * 3 + 0];
    float ry = x[s * 3 + 1] - x[d * 3 + 1];
    float rz = x[s * 3 + 2] - x[d * 3 + 2];
    float d2 = rx * rx + ry * ry + rz * rz;
    float w = sv / (sqrtf(d2) + 1.f);
    atomicAdd(&dx[d * 3 + 0], w * rx);
    atomicAdd(&dx[d * 3 + 1], w * ry);
    atomicAdd(&dx[d * 3 + 2], w * rz);
  }
}

// ---------------- h = LN(h + tmp) ; one wave per node; also emit bf16 h ----------
__global__ void k_ln(float* __restrict__ h, const float* __restrict__ tmp,
                     const float* __restrict__ g, const float* __restrict__ b,
                     ushort_t* __restrict__ hb) {
  int node = blockIdx.x * 8 + (threadIdx.x >> 5);
  int lane = threadIdx.x & 31;
  int c0 = lane * 4;
  float4 a = *reinterpret_cast<const float4*>(h + (size_t)node * CC + c0);
  float4 t = *reinterpret_cast<const float4*>(tmp + (size_t)node * CC + c0);
  float v0 = a.x + t.x, v1 = a.y + t.y, v2 = a.z + t.z, v3 = a.w + t.w;
  float s = v0 + v1 + v2 + v3;
#pragma unroll
  for (int m = 1; m <= 16; m <<= 1) s += __shfl_xor(s, m, 32);
  float mean = s * (1.f / 128.f);
  float q = (v0 - mean) * (v0 - mean) + (v1 - mean) * (v1 - mean) +
            (v2 - mean) * (v2 - mean) + (v3 - mean) * (v3 - mean);
#pragma unroll
  for (int m = 1; m <= 16; m <<= 1) q += __shfl_xor(q, m, 32);
  float invs = rsqrtf(q * (1.f / 128.f) + 1e-5f);
  float o0 = (v0 - mean) * invs * g[c0 + 0] + b[c0 + 0];
  float o1 = (v1 - mean) * invs * g[c0 + 1] + b[c0 + 1];
  float o2 = (v2 - mean) * invs * g[c0 + 2] + b[c0 + 2];
  float o3 = (v3 - mean) * invs * g[c0 + 3] + b[c0 + 3];
  float* hp = h + (size_t)node * CC + c0;
  hp[0] = o0; hp[1] = o1; hp[2] = o2; hp[3] = o3;
  uint2 o;
  o.x = bf16_2(o0, o1);
  o.y = bf16_2(o2, o3);
  *reinterpret_cast<uint2*>(hb + (size_t)node * CC + c0) = o;
}

__global__ void k_xadd(float* __restrict__ x, const float* __restrict__ dx) {
  int i = blockIdx.x * 256 + threadIdx.x;
  if (i < NNODE * 3) x[i] += dx[i];
}

// ---------------- host orchestration ----------------
extern "C" void kernel_launch(void* const* d_in, const int* in_sizes, int n_in,
                              void* d_out, int out_size, void* d_ws, size_t ws_size,
                              hipStream_t stream) {
  const float* seq    = (const float*)d_in[0];
  const float* pair   = (const float*)d_in[1];
  const float* bppm   = (const float*)d_in[2];
  const float* coords = (const float*)d_in[3];
  const float* W_in   = (const float*)d_in[4];
  const float* Wq     = (const float*)d_in[5];
  const float* Wk     = (const float*)d_in[6];
  const float* Wv     = (const float*)d_in[7];
  const float* Wo     = (const float*)d_in[8];
  const float* We     = (const float*)d_in[9];
  const float* wd     = (const float*)d_in[10];
  const float* wx     = (const float*)d_in[11];
  const float* lng    = (const float*)d_in[12];
  const float* lnb    = (const float*)d_in[13];
  const int*   src    = (const int*)d_in[14];
  const int*   dst    = (const int*)d_in[15];
  const float* emask  = (const float*)d_in[16];
  (void)in_sizes; (void)n_in; (void)out_size; (void)ws_size;

  char* ws = (char*)d_ws;
  size_t off = 0;
  auto alloc = [&](size_t bytes) -> void* {
    void* p = ws + off;
    off = (off + bytes + 255) & ~(size_t)255;
    return p;
  };
  ushort_t* ebf  = (ushort_t*)alloc((size_t)EMAX * CC * 2);
  float* bpe     = (float*)alloc((size_t)EMAX * 4);
  float* h       = (float*)alloc((size_t)NNODE * CC * 4);
  float* hq      = (float*)alloc((size_t)NNODE * CC * 4);
  float* hk      = (float*)alloc((size_t)NNODE * CC * 4);
  float* hv      = (float*)alloc((size_t)NNODE * CC * 4);
  float* tmp     = (float*)alloc((size_t)NNODE * CC * 4);
  float* logits  = (float*)alloc((size_t)EMAX * HH * 4);
  float* aex     = (float*)alloc((size_t)EMAX * HH * 4);
  float* sdot    = (float*)alloc((size_t)EMAX * 4);
  unsigned* nmax = (unsigned*)alloc((size_t)NNODE * HH * 4);
  float* den     = (float*)alloc((size_t)NNODE * HH * 4);
  float* agg     = (float*)alloc((size_t)NNODE * CC * 4);
  float* dx      = (float*)alloc((size_t)NNODE * 3 * 4);
  float* xws     = (float*)alloc((size_t)NNODE * 3 * 4);
  // bf16 activations / pre-transposed bf16 weights (N-major B-operand layout)
  ushort_t* seqb = (ushort_t*)alloc((size_t)NNODE * SD * 2);
  ushort_t* hb   = (ushort_t*)alloc((size_t)NNODE * CC * 2);
  ushort_t* aggb = (ushort_t*)alloc((size_t)NNODE * CC * 2);
  ushort_t* WbI  = (ushort_t*)alloc((size_t)128 * SD * 2);
  ushort_t* WbQ  = (ushort_t*)alloc((size_t)NLAY * 128 * 128 * 2);
  ushort_t* WbK  = (ushort_t*)alloc((size_t)NLAY * 128 * 128 * 2);
  ushort_t* WbV  = (ushort_t*)alloc((size_t)NLAY * 128 * 128 * 2);
  ushort_t* WbO  = (ushort_t*)alloc((size_t)NLAY * 128 * 128 * 2);
  ushort_t* WbE  = (ushort_t*)alloc((size_t)NLAY * 128 * 128 * 2);

  hipMemcpyAsync(xws, coords, (size_t)NNODE * 3 * sizeof(float),
                 hipMemcpyDeviceToDevice, stream);
  k_edgefeat<<<EMAX / 8, 256, 0, stream>>>(pair, bppm, src, dst, ebf, bpe);
  k_f2bf<<<(NNODE * SD + 255) / 256, 256, 0, stream>>>(seq, seqb, NNODE * SD);
  k_w2bf<<<(SD * 128 + 255) / 256, 256, 0, stream>>>(W_in, WbI, SD);
  for (int l = 0; l < NLAY; ++l) {
    k_w2bf<<<64, 256, 0, stream>>>(Wq + (size_t)l * CC * CC, WbQ + (size_t)l * CC * CC, CC);
    k_w2bf<<<64, 256, 0, stream>>>(Wk + (size_t)l * CC * CC, WbK + (size_t)l * CC * CC, CC);
    k_w2bf<<<64, 256, 0, stream>>>(Wv + (size_t)l * CC * CC, WbV + (size_t)l * CC * CC, CC);
    k_w2bf<<<64, 256, 0, stream>>>(Wo + (size_t)l * CC * CC, WbO + (size_t)l * CC * CC, CC);
    k_w2bf<<<64, 256, 0, stream>>>(We + (size_t)l * 129 * CC, WbE + (size_t)l * CC * CC, CC);
  }
  // h = seq @ W_in  (2048 x 640 x 128)
  k_gemm<<<NNODE / 64, 128, 0, stream>>>(seqb, WbI, h, NNODE, SD, 0);
  k_f2bf<<<(NNODE * CC + 255) / 256, 256, 0, stream>>>(h, hb, NNODE * CC);

  for (int l = 0; l < NLAY; ++l) {
    k_gemm<<<NNODE / 64, 128, 0, stream>>>(hb, WbQ + (size_t)l * CC * CC, hq, NNODE, CC, 0);
    k_gemm<<<NNODE / 64, 128, 0, stream>>>(hb, WbK + (size_t)l * CC * CC, hk, NNODE, CC, 0);
    k_gemm<<<NNODE / 64, 128, 0, stream>>>(hb, WbV + (size_t)l * CC * CC, hv, NNODE, CC, 0);
    k_init<<<NNODE * CC / 256, 256, 0, stream>>>(den, agg, dx, nmax);
    k_edge<<<256, 256, 0, stream>>>(ebf, bpe, hq, hk, xws, src, dst, emask,
                                    WbE + (size_t)l * CC * CC,
                                    We + (size_t)l * 129 * CC, wd + (size_t)l * CC,
                                    wx + (size_t)l * CC, logits, sdot);
    k_smax<<<EMAX * HH / 256, 256, 0, stream>>>(logits, dst, nmax);
    k_sexp<<<EMAX * HH / 256, 256, 0, stream>>>(logits, dst, nmax, aex, den);
    k_agg<<<EMAX / 8, 256, 0, stream>>>(aex, den, hv, sdot, xws, src, dst, emask, agg, dx);
    k_f2bf<<<(NNODE * CC + 255) / 256, 256, 0, stream>>>(agg, aggb, NNODE * CC);
    k_gemm<<<NNODE / 64, 128, 0, stream>>>(aggb, WbO + (size_t)l * CC * CC, tmp, NNODE, CC, 1);
    k_ln<<<NNODE / 8, 256, 0, stream>>>(h, tmp, lng + (size_t)l * CC, lnb + (size_t)l * CC, hb);
    k_xadd<<<(NNODE * 3 + 255) / 256, 256, 0, stream>>>(xws, dx);
  }
  hipMemcpyAsync(d_out, xws, (size_t)NNODE * 3 * sizeof(float),
                 hipMemcpyDeviceToDevice, stream);
}
